// MPNNWithAttention_57011395887341
// MI455X (gfx1250) — matrix operand order repaired and checked by reading, not compiled, not run on hardware
//
#include <hip/hip_runtime.h>
#include <hip/hip_bf16.h>

// ---------------------------------------------------------------------------
// GAT (3-layer, 4-head, concat=False) + mean-pool + FC for MI455X (gfx1250).
// - GEMMs: v_wmma_f32_16x16x32_f16 (f16 in, f32 accumulate), b128 direct loads.
// - Edge softmax/aggregation: f16 z/agg (L2-resident, 25.6 MB each),
//   global_atomic_pk_add_f16 packed atomics, global_prefetch_b8 lookahead.
// ---------------------------------------------------------------------------

namespace {
constexpr int kN      = 50000;   // nodes
constexpr int kE      = 800000;  // edges (self loops handled separately)
constexpr int kIN     = 128;
constexpr int kH      = 64;
constexpr int kHeads  = 4;
constexpr int kLayers = 3;
constexpr int kG      = 256;
constexpr int kOut    = 326000;
constexpr int kHH     = kHeads * kH;  // 256
constexpr float kNeg  = 0.2f;
}

typedef __attribute__((ext_vector_type(16))) _Float16 v16h;
typedef __attribute__((ext_vector_type(8)))  _Float16 v8h;
typedef __attribute__((ext_vector_type(2)))  _Float16 v2h;
typedef __attribute__((ext_vector_type(8)))  float    v8f;

__device__ __forceinline__ float lrelu(float x) { return x > 0.f ? x : kNeg * x; }

// float atomic max via ordered integer encoding (works for mixed signs)
__device__ __forceinline__ void atomicMaxF(float* a, float v) {
  if (v >= 0.f) atomicMax((int*)a, __float_as_int(v));
  else          atomicMin((unsigned int*)a, __float_as_uint(v));
}

// CDNA5 packed-f16 atomic add (GLOBAL_ATOMIC_PK_ADD_F16, no return)
__device__ __forceinline__ void atomicPkAddF16(_Float16* addr, v2h val) {
  asm volatile("global_atomic_pk_add_f16 %0, %1, off"
               :: "v"(addr), "v"(val) : "memory");
}

// --------------------------------- casts -----------------------------------

__global__ void k_cast_f16(const float* __restrict__ s, _Float16* __restrict__ d,
                           long n) {
  long i = (long)blockIdx.x * blockDim.x + threadIdx.x;
  if (i < n) d[i] = (_Float16)s[i];
}

__global__ void k_zero_f32(float* __restrict__ p, int n) {
  int i = blockIdx.x * blockDim.x + threadIdx.x;
  if (i < n) p[i] = 0.f;
}

// ------------------------------ WMMA GEMM ----------------------------------
// C[M,N] = A[M,K] @ B[N,K]^T (+ bias[N]) (+ relu). M%16==0, N%16==0, K%32==0.
// Optional f32 output C and/or f16 output C16.
// Block = 256 threads = 8 waves; wave w owns output tile (blockIdx.y,
// blockIdx.x*8+w). Each wave: 16x16 f32 tile, K-loop step 32.
__global__ __launch_bounds__(256)
void k_wmma_gemm(const _Float16* __restrict__ A, const _Float16* __restrict__ B,
                 const float* __restrict__ bias, float* __restrict__ C,
                 _Float16* __restrict__ C16, int M, int N, int K, int act) {
  const int lane = threadIdx.x & 31;
  const int wv   = threadIdx.x >> 5;
  const int nt   = blockIdx.x * 8 + wv;
  const int mt   = blockIdx.y;
  if (nt * 16 >= N) return;

  const int half = lane >> 4;    // which 16-lane group
  const int l16  = lane & 15;

  // A fragment: lane row = mt*16+l16; half 0 -> K {0..7, 16..23}, half 1 -> +8
  const _Float16* Arow = A + (size_t)(mt * 16 + l16) * K;
  // B fragment: lane col = nt*16+l16; half 0 -> K {0..7, 16..23}, half 1 -> +8
  const _Float16* Brow = B + (size_t)(nt * 16 + l16) * K;

  v8f acc = {};
  for (int k0 = 0; k0 < K; k0 += 32) {
    v8h alo = *(const v8h*)(Arow + k0 + half * 8);
    v8h ahi = *(const v8h*)(Arow + k0 + half * 8 + 16);
    v8h b0  = *(const v8h*)(Brow + k0 + half * 8);
    v8h b1  = *(const v8h*)(Brow + k0 + half * 8 + 16);
    v16h a, b;
#pragma unroll
    for (int e = 0; e < 8; ++e) {
      a[e] = alo[e]; a[e + 8] = ahi[e];
      b[e] = b0[e];  b[e + 8] = b1[e];
    }
    acc = __builtin_amdgcn_wmma_f32_16x16x32_f16(false, a, false, b,
                                                 (short)0, acc, false, false);
  }

  const int ocol = nt * 16 + l16;
  const float bv = bias ? bias[ocol] : 0.f;
#pragma unroll
  for (int r = 0; r < 8; ++r) {          // acc[r]: row r + 8*half, col l16
    int orow = mt * 16 + half * 8 + r;
    float v = acc[r] + bv;
    if (act) v = fmaxf(v, 0.f);
    size_t idx = (size_t)orow * N + ocol;
    if (C)   C[idx]   = v;
    if (C16) C16[idx] = (_Float16)v;
  }
}

// --------------------------- attention scores ------------------------------
// a_s[n,h] = <z[n,h,:], att_src[h,:]>, a_d likewise. z is f16.
__global__ void k_att(const _Float16* __restrict__ z, const float* __restrict__ asrc,
                      const float* __restrict__ adst, float* __restrict__ a_s,
                      float* __restrict__ a_d) {
  int i = blockIdx.x * blockDim.x + threadIdx.x;
  if (i >= kN * kHeads) return;
  int n = i >> 2, h = i & 3;
  const _Float16* zp = z + (size_t)n * kHH + h * kH;
  const float* ps = asrc + h * kH;
  const float* pd = adst + h * kH;
  float ss = 0.f, sd = 0.f;
#pragma unroll 8
  for (int d = 0; d < kH; ++d) {
    float v = (float)zp[d];
    ss += v * ps[d]; sd += v * pd[d];
  }
  a_s[i] = ss;
  a_d[i] = sd;
}

// --------------------------- softmax over edges ----------------------------

// init max with the self-loop score (self loop always present)
__global__ void k_init_m(const float* __restrict__ a_s, const float* __restrict__ a_d,
                         float* __restrict__ m) {
  int i = blockIdx.x * blockDim.x + threadIdx.x;
  if (i >= kN * kHeads) return;
  m[i] = lrelu(a_s[i] + a_d[i]);
}

__global__ void k_edge_max(const int* __restrict__ ei, const float* __restrict__ a_s,
                           const float* __restrict__ a_d, float* __restrict__ m) {
  int i = blockIdx.x * blockDim.x + threadIdx.x;
  if (i >= kE * kHeads) return;
  int e = i >> 2, h = i & 3;
  int s = ei[e], d = ei[kE + e];
  float v = lrelu(a_s[s * kHeads + h] + a_d[d * kHeads + h]);
  atomicMaxF(&m[d * kHeads + h], v);
}

__global__ void k_self_denom(const float* __restrict__ a_s, const float* __restrict__ a_d,
                             const float* __restrict__ m, float* __restrict__ denom) {
  int i = blockIdx.x * blockDim.x + threadIdx.x;
  if (i >= kN * kHeads) return;
  denom[i] = __expf(lrelu(a_s[i] + a_d[i]) - m[i]);
}

__global__ void k_edge_denom(const int* __restrict__ ei, const float* __restrict__ a_s,
                             const float* __restrict__ a_d, const float* __restrict__ m,
                             float* __restrict__ denom) {
  int i = blockIdx.x * blockDim.x + threadIdx.x;
  if (i >= kE * kHeads) return;
  int e = i >> 2, h = i & 3;
  int s = ei[e], d = ei[kE + e];
  float v = lrelu(a_s[s * kHeads + h] + a_d[d * kHeads + h]);
  atomicAdd(&denom[d * kHeads + h], __expf(v - m[d * kHeads + h]));
}

// initialize agg with the self-loop message (full overwrite -> no memset)
__global__ void k_self_msg(const _Float16* __restrict__ z, const float* __restrict__ a_s,
                           const float* __restrict__ a_d, const float* __restrict__ m,
                           const float* __restrict__ denom, _Float16* __restrict__ agg) {
  long i = (long)blockIdx.x * blockDim.x + threadIdx.x;   // half2 index
  if (i >= (long)kN * (kHH / 2)) return;
  int n = (int)(i >> 7), c2 = (int)(i & 127), h = c2 >> 5;
  float se = lrelu(a_s[n * kHeads + h] + a_d[n * kHeads + h]);
  float alpha = __expf(se - m[n * kHeads + h]) / denom[n * kHeads + h];
  v2h z2 = *(const v2h*)(z + (size_t)n * kHH + 2 * c2);
  v2h o;
  o[0] = (_Float16)((float)z2[0] * alpha);
  o[1] = (_Float16)((float)z2[1] * alpha);
  *(v2h*)(agg + (size_t)n * kHH + 2 * c2) = o;
}

// 128 threads per edge (one half2/thread, one wave per head); grid-stride with
// prefetch of the next edge's source row; packed f16 atomics into agg.
__global__ __launch_bounds__(128)
void k_edge_msg(const int* __restrict__ ei, const _Float16* __restrict__ z,
                const float* __restrict__ a_s, const float* __restrict__ a_d,
                const float* __restrict__ m, const float* __restrict__ denom,
                _Float16* __restrict__ agg) {
  const int c2 = threadIdx.x;       // half2 index 0..127
  const int h  = c2 >> 5;           // wave id == head id
  for (int e = blockIdx.x; e < kE; e += (int)gridDim.x) {
    int en = e + (int)gridDim.x;
    if (en < kE) {                  // prefetch next iteration's z row
      int sn = ei[en];
      __builtin_prefetch(z + (size_t)sn * kHH + 2 * c2, 0, 0);
    }
    int s = ei[e], d = ei[kE + e];
    float v = lrelu(a_s[s * kHeads + h] + a_d[d * kHeads + h]);
    float alpha = __expf(v - m[d * kHeads + h]) / denom[d * kHeads + h];
    v2h z2 = *(const v2h*)(z + (size_t)s * kHH + 2 * c2);
    v2h o;
    o[0] = (_Float16)((float)z2[0] * alpha);
    o[1] = (_Float16)((float)z2[1] * alpha);
    atomicPkAddF16(agg + (size_t)d * kHH + 2 * c2, o);
  }
}

// mean over heads + bias + relu; produce f32 (pooling) and f16 (next GEMM)
__global__ void k_finish(const _Float16* __restrict__ agg, const float* __restrict__ gb,
                         float* __restrict__ h, _Float16* __restrict__ h16) {
  long i = (long)blockIdx.x * blockDim.x + threadIdx.x;
  if (i >= (long)kN * kH) return;
  int n = (int)(i >> 6), d = (int)(i & 63);
  const _Float16* ap = agg + (size_t)n * kHH;
  float v = 0.25f * ((float)ap[d] + (float)ap[kH + d] +
                     (float)ap[2 * kH + d] + (float)ap[3 * kH + d]) + gb[d];
  v = fmaxf(v, 0.f);
  h[i] = v;
  h16[i] = (_Float16)v;
}

// ------------------------------- pooling -----------------------------------

__global__ void k_pool(const int* __restrict__ batch, const float* __restrict__ h,
                       float* __restrict__ gsum, float* __restrict__ cnt) {
  long i = (long)blockIdx.x * blockDim.x + threadIdx.x;
  if (i >= (long)kN * kH) return;
  int n = (int)(i >> 6), d = (int)(i & 63);
  int g = batch[n];
  atomicAdd(&gsum[g * kH + d], h[i]);
  if (d == 0) atomicAdd(&cnt[g], 1.f);
}

__global__ void k_pooled(const float* __restrict__ gsum, const float* __restrict__ cnt,
                         _Float16* __restrict__ p16) {
  int i = blockIdx.x * blockDim.x + threadIdx.x;
  if (i >= kG * kH) return;
  float p = gsum[i] / fmaxf(cnt[i >> 6], 1.f);
  p16[i] = (_Float16)p;
}

// ---------------------------------------------------------------------------

extern "C" void kernel_launch(void* const* d_in, const int* in_sizes, int n_in,
                              void* d_out, int out_size, void* d_ws, size_t ws_size,
                              hipStream_t stream) {
  const float* x       = (const float*)d_in[0];
  const int*   ei      = (const int*)d_in[1];
  // d_in[2] edge_attr: unused by reference
  const int*   batch   = (const int*)d_in[3];
  const float* emb_w   = (const float*)d_in[4];
  const float* emb_b   = (const float*)d_in[5];
  const float* lin_w   = (const float*)d_in[6];
  const float* att_src = (const float*)d_in[7];
  const float* att_dst = (const float*)d_in[8];
  const float* gat_b   = (const float*)d_in[9];
  const float* fc_w    = (const float*)d_in[10];
  const float* fc_b    = (const float*)d_in[11];
  float* out = (float*)d_out;

  char* ws = (char*)d_ws;
  size_t off = 0;
  auto alloc = [&](size_t bytes) -> char* {
    char* p = ws + off;
    off = (off + bytes + 255) & ~(size_t)255;
    return p;
  };

  _Float16* x16  = (_Float16*)alloc((size_t)kN * kIN * sizeof(_Float16));
  _Float16* ew16 = (_Float16*)alloc((size_t)kH * kIN * sizeof(_Float16));
  _Float16* lw16 = (_Float16*)alloc((size_t)kLayers * kHH * kH * sizeof(_Float16));
  _Float16* fw16 = (_Float16*)alloc((size_t)kOut * kH * sizeof(_Float16));
  float*    hbuf = (float*)alloc((size_t)kN * kH * sizeof(float));
  _Float16* h16  = (_Float16*)alloc((size_t)kN * kH * sizeof(_Float16));
  _Float16* z16  = (_Float16*)alloc((size_t)kN * kHH * sizeof(_Float16));
  float*    a_s  = (float*)alloc((size_t)kN * kHeads * sizeof(float));
  float*    a_d  = (float*)alloc((size_t)kN * kHeads * sizeof(float));
  float*    mmax = (float*)alloc((size_t)kN * kHeads * sizeof(float));
  float*    den  = (float*)alloc((size_t)kN * kHeads * sizeof(float));
  _Float16* agg16= (_Float16*)alloc((size_t)kN * kHH * sizeof(_Float16));
  float*    gsum = (float*)alloc((size_t)kG * kH * sizeof(float));
  float*    cnt  = (float*)alloc((size_t)kG * sizeof(float));
  _Float16* p16  = (_Float16*)alloc((size_t)kG * kH * sizeof(_Float16));
  (void)ws_size; (void)n_in; (void)in_sizes; (void)out_size;

  const int T = 256;
  auto blk1 = [&](long n) { return dim3((unsigned)((n + T - 1) / T)); };

  // f16 copies of inputs/weights
  k_cast_f16<<<blk1((long)kN * kIN), T, 0, stream>>>(x, x16, (long)kN * kIN);
  k_cast_f16<<<blk1(kH * kIN), T, 0, stream>>>(emb_w, ew16, kH * kIN);
  k_cast_f16<<<blk1(kLayers * kHH * kH), T, 0, stream>>>(lin_w, lw16,
                                                         kLayers * kHH * kH);
  k_cast_f16<<<blk1((long)kOut * kH), T, 0, stream>>>(fc_w, fw16, (long)kOut * kH);

  // node embedding: h16 = f16(x @ emb_w^T + emb_b)   [50000,64]
  {
    dim3 grid((kH / 16 + 7) / 8, kN / 16);
    k_wmma_gemm<<<grid, T, 0, stream>>>(x16, ew16, emb_b, nullptr, h16,
                                        kN, kH, kIN, 0);
  }

  for (int l = 0; l < kLayers; ++l) {
    // z16 = f16(h @ lin_w[l]^T)   [50000,256]
    dim3 grid((kHH / 16 + 7) / 8, kN / 16);
    k_wmma_gemm<<<grid, T, 0, stream>>>(h16, lw16 + (size_t)l * kHH * kH, nullptr,
                                        nullptr, z16, kN, kHH, kH, 0);

    k_att<<<blk1(kN * kHeads), T, 0, stream>>>(z16, att_src + l * kHeads * kH,
                                               att_dst + l * kHeads * kH, a_s, a_d);
    k_init_m<<<blk1(kN * kHeads), T, 0, stream>>>(a_s, a_d, mmax);
    k_edge_max<<<blk1((long)kE * kHeads), T, 0, stream>>>(ei, a_s, a_d, mmax);
    k_self_denom<<<blk1(kN * kHeads), T, 0, stream>>>(a_s, a_d, mmax, den);
    k_edge_denom<<<blk1((long)kE * kHeads), T, 0, stream>>>(ei, a_s, a_d, mmax, den);
    k_self_msg<<<blk1((long)kN * (kHH / 2)), T, 0, stream>>>(z16, a_s, a_d, mmax,
                                                             den, agg16);
    k_edge_msg<<<dim3(32768), 128, 0, stream>>>(ei, z16, a_s, a_d, mmax, den, agg16);
    k_finish<<<blk1((long)kN * kH), T, 0, stream>>>(agg16, gat_b + l * kH, hbuf, h16);
  }

  // global mean pool
  k_zero_f32<<<blk1(kG * kH), T, 0, stream>>>(gsum, kG * kH);
  k_zero_f32<<<blk1(kG), T, 0, stream>>>(cnt, kG);
  k_pool<<<blk1((long)kN * kH), T, 0, stream>>>(batch, hbuf, gsum, cnt);
  k_pooled<<<blk1(kG * kH), T, 0, stream>>>(gsum, cnt, p16);

  // out = pooled @ fc_w^T + fc_b   [256, 326000]
  {
    dim3 grid((kOut / 16 + 7) / 8, kG / 16);
    k_wmma_gemm<<<grid, T, 0, stream>>>(p16, fw16, fc_b, out, nullptr,
                                        kG, kOut, kH, 0);
  }
}
